// RGLRU_40441412059391
// MI455X (gfx1250) — matrix operand order repaired and checked by reading, not compiled, not run on hardware
//
#include <hip/hip_runtime.h>
#include <hip/hip_bf16.h>
#include <math.h>

// ---------------------------------------------------------------------------
// RG-LRU for MI455X (gfx1250): bf16 WMMA GEMMs with async-to-LDS double
// buffering + packed weights + fused gate epilogue + diagonal scan
// ---------------------------------------------------------------------------

typedef __bf16 bf16_t;
typedef __attribute__((ext_vector_type(16))) __bf16 v16bf;
typedef __attribute__((ext_vector_type(8)))  __bf16 v8bf;
typedef __attribute__((ext_vector_type(8)))  float  v8f;
typedef __attribute__((ext_vector_type(4)))  float  v4f;

#define D_DIM 1024
#define S_DIM 8192
#define B_DIM 8
#define M_TOT (B_DIM * S_DIM)   // 65536 rows
#define KSTEP 32                // bf16 WMMA K per stage
#define NSTG  (D_DIM / KSTEP)   // 32 K-stages
#define LDSK  40                // padded LDS k-stride (elements): 80B rows, 16B-aligned chunks

// ---- CDNA5 async global->LDS copy (ASYNCcnt-tracked), 16 bytes per lane ----
__device__ __forceinline__ void async_b128(unsigned int lds_off, const void* g) {
    asm volatile("global_load_async_to_lds_b128 %0, %1, off"
                 :: "v"(lds_off), "v"((unsigned long long)(uintptr_t)g)
                 : "memory");
}
__device__ __forceinline__ unsigned int lds_off(const void* p) {
    // Generic pointers to LDS carry the LDS byte offset in the low 32 bits.
    return (unsigned int)(uintptr_t)p;
}
__device__ __forceinline__ void wait_async0() {
#if __has_builtin(__builtin_amdgcn_s_wait_asynccnt)
    __builtin_amdgcn_s_wait_asynccnt(0);
#else
    asm volatile("s_wait_asynccnt 0x0" ::: "memory");
#endif
}

// Build a 16-element bf16 fragment from two 16-byte LDS chunks.
__device__ __forceinline__ v16bf make_frag(const bf16_t* p0, const bf16_t* p1) {
    v8bf lo = *(const v8bf*)p0;
    v8bf hi = *(const v8bf*)p1;
    return __builtin_shufflevector(lo, hi, 0,1,2,3,4,5,6,7,8,9,10,11,12,13,14,15);
}

// ===========================================================================
// Weight pack: W[k][n] f32 -> Wp[n][k] bf16 (transposed, contiguous K rows).
// Makes every B-tile stage a pure byte copy (async-to-LDS eligible).
// ===========================================================================
__global__ __launch_bounds__(256) void pack_w(const float* __restrict__ W,
                                              bf16_t* __restrict__ Wp)
{
    __shared__ float t[32][33];
    const int k0 = blockIdx.y * 32;
    const int n0 = blockIdx.x * 32;
    {
        const int r = threadIdx.x >> 3;           // 0..31 (k)
        const int c = (threadIdx.x & 7) * 4;      // 0..28 (n)
        v4f v = *(const v4f*)(W + (size_t)(k0 + r) * D_DIM + n0 + c);
        t[r][c + 0] = v.x; t[r][c + 1] = v.y; t[r][c + 2] = v.z; t[r][c + 3] = v.w;
    }
    __syncthreads();
    {
        const int n = threadIdx.x >> 3;           // 0..31 (n)
        const int k = (threadIdx.x & 7) * 4;      // 0..28 (k)
        bf16_t* o = Wp + (size_t)(n0 + n) * D_DIM + k0 + k;
        o[0] = (bf16_t)t[k + 0][n]; o[1] = (bf16_t)t[k + 1][n];
        o[2] = (bf16_t)t[k + 2][n]; o[3] = (bf16_t)t[k + 3][n];
    }
}

// ===========================================================================
// Single-B GEMM: C[M,1024] = A[M,1024] @ W[1024,1024] + bias
// MODE 0: A = f32 global (x) [register-staged convert], Out = bf16 (x1)
// MODE 1: A = bf16 scratch (h) [async-to-LDS], Out = f32 (d_out)
// Workgroup tile 128x128, 8 waves as 2x4, wave tile 64x32 (4x2 WMMA frags).
// Double-buffered LDS; B always async from packed bf16 weights.
// ===========================================================================
template<int MODE>
__global__ __launch_bounds__(256) void gemm_single(
    const void* __restrict__ Ap, const bf16_t* __restrict__ Wp,
    const float* __restrict__ bias, void* __restrict__ Out)
{
    __shared__ bf16_t lA[2][128 * LDSK];
    __shared__ bf16_t lB[2][128 * LDSK];

    const int tid  = threadIdx.x;
    const int lane = tid & 31;
    const int wv   = tid >> 5;
    const int half = lane >> 4;
    const int l16  = lane & 15;
    const int wr   = wv >> 2;        // 0..1 (M)
    const int wc   = wv & 3;         // 0..3 (N)
    const int m0   = blockIdx.y * 128;
    const int n0   = blockIdx.x * 128;

    const int arow = tid >> 1;           // 0..127 (MODE 0 fill)
    const int acol = (tid & 1) * 16;     // 0 or 16

    v8f acc[4][2] = {};
    v4f ar[4];                           // MODE 0 register staging

    // ---- fill helpers ----
    auto fillB = [&](int s, int buf) {
        #pragma unroll
        for (int j = 0; j < 2; ++j) {
            const int c = tid * 2 + j;           // 0..511
            const int n = c >> 2, part = c & 3;  // 128 rows x 4 x 16B
            async_b128(lds_off(&lB[buf][n * LDSK + part * 8]),
                       Wp + (size_t)(n0 + n) * D_DIM + s * KSTEP + part * 8);
        }
    };
    auto fillA_async = [&](int s, int buf) {     // MODE 1: bf16 A
        #pragma unroll
        for (int j = 0; j < 2; ++j) {
            const int c = tid * 2 + j;
            const int r = c >> 2, part = c & 3;
            async_b128(lds_off(&lA[buf][r * LDSK + part * 8]),
                       (const bf16_t*)Ap + (size_t)(m0 + r) * D_DIM + s * KSTEP + part * 8);
        }
    };
    auto gloadA = [&](int s) {                   // MODE 0: f32 A -> regs
        const float* A = (const float*)Ap + (size_t)(m0 + arow) * D_DIM + s * KSTEP + acol;
        #pragma unroll
        for (int j = 0; j < 4; ++j) ar[j] = *(const v4f*)(A + j * 4);
    };
    auto storeA = [&](int buf) {                 // regs -> LDS as bf16
        v8bf o0, o1;
        o0[0] = (bf16_t)ar[0].x; o0[1] = (bf16_t)ar[0].y; o0[2] = (bf16_t)ar[0].z; o0[3] = (bf16_t)ar[0].w;
        o0[4] = (bf16_t)ar[1].x; o0[5] = (bf16_t)ar[1].y; o0[6] = (bf16_t)ar[1].z; o0[7] = (bf16_t)ar[1].w;
        o1[0] = (bf16_t)ar[2].x; o1[1] = (bf16_t)ar[2].y; o1[2] = (bf16_t)ar[2].z; o1[3] = (bf16_t)ar[2].w;
        o1[4] = (bf16_t)ar[3].x; o1[5] = (bf16_t)ar[3].y; o1[6] = (bf16_t)ar[3].z; o1[7] = (bf16_t)ar[3].w;
        *(v8bf*)&lA[buf][arow * LDSK + acol]     = o0;
        *(v8bf*)&lA[buf][arow * LDSK + acol + 8] = o1;
    };
    auto compute = [&](int buf) {
        v16bf bfr[2];
        #pragma unroll
        for (int ni = 0; ni < 2; ++ni) {
            const bf16_t* bp = &lB[buf][(wc * 32 + ni * 16 + l16) * LDSK + half * 8];
            bfr[ni] = make_frag(bp, bp + 16);
        }
        #pragma unroll
        for (int mi = 0; mi < 4; ++mi) {
            const bf16_t* ap = &lA[buf][(wr * 64 + mi * 16 + l16) * LDSK + half * 8];
            v16bf afr = make_frag(ap, ap + 16);
            #pragma unroll
            for (int ni = 0; ni < 2; ++ni) {
                acc[mi][ni] = __builtin_amdgcn_wmma_f32_16x16x32_bf16(
                    false, afr, false, bfr[ni], (short)0, acc[mi][ni], false, false);
            }
        }
    };

    // ---- prologue: stage 0 into buffer 0 ----
    fillB(0, 0);
    if (MODE == 0) { gloadA(0); storeA(0); }
    else           { fillA_async(0, 0); }
    wait_async0();
    __syncthreads();

    // ---- double-buffered main loop ----
    for (int s = 0; s < NSTG; ++s) {
        const int  cur  = s & 1;
        const bool more = (s + 1 < NSTG);
        if (more) {
            fillB(s + 1, cur ^ 1);
            if (MODE == 0) gloadA(s + 1);
            else           fillA_async(s + 1, cur ^ 1);
        }
        compute(cur);
        if (more && MODE == 0) storeA(cur ^ 1);
        wait_async0();
        __syncthreads();
    }

    // ---- epilogue ----
    #pragma unroll
    for (int ni = 0; ni < 2; ++ni) {
        const int n = n0 + wc * 32 + ni * 16 + l16;
        const float bb = bias[n];
        #pragma unroll
        for (int mi = 0; mi < 4; ++mi) {
            #pragma unroll
            for (int r = 0; r < 8; ++r) {
                const int m = m0 + wr * 64 + mi * 16 + half * 8 + r;
                const float v = acc[mi][ni][r] + bb;
                if (MODE == 0)
                    ((bf16_t*)Out)[(size_t)m * D_DIM + n] = (bf16_t)v;
                else
                    ((float*)Out)[(size_t)m * D_DIM + n] = v;
            }
        }
    }
}

// ===========================================================================
// Dual-B GEMM + fused RG-LRU gate epilogue:
//   gx = sigmoid(x1@W_gx + b_gx), ga = sigmoid(x1@W_ga + b_ga)
//   log_a = -8*ga*softplus(a_param); a = exp(log_a); sc = sqrt(1-exp(2log_a))
//   writes a (f32) and y = sc*gx*x1 (bf16)
// Workgroup tile 64x128, 8 waves as 2x4, wave tile 32x32 per output.
// Fully async-to-LDS, double-buffered.
// ===========================================================================
__global__ __launch_bounds__(256) void gemm_dual(
    const bf16_t* __restrict__ X1, const bf16_t* __restrict__ Wpx,
    const bf16_t* __restrict__ Wpa, const float* __restrict__ bgx,
    const float* __restrict__ bga, const float* __restrict__ aparam,
    float* __restrict__ aOut, bf16_t* __restrict__ yOut)
{
    __shared__ bf16_t lA [2][64  * LDSK];
    __shared__ bf16_t lBx[2][128 * LDSK];
    __shared__ bf16_t lBa[2][128 * LDSK];

    const int tid  = threadIdx.x;
    const int lane = tid & 31;
    const int wv   = tid >> 5;
    const int half = lane >> 4;
    const int l16  = lane & 15;
    const int wr   = wv >> 2;        // 0..1 (M)
    const int wc   = wv & 3;         // 0..3 (N)
    const int m0   = blockIdx.y * 64;
    const int n0   = blockIdx.x * 128;

    v8f accx[2][2] = {};
    v8f acca[2][2] = {};

    auto fill = [&](int s, int buf) {
        {   // A: 64 rows x 4 chunks = 256, one per thread
            const int r = tid >> 2, part = tid & 3;
            async_b128(lds_off(&lA[buf][r * LDSK + part * 8]),
                       X1 + (size_t)(m0 + r) * D_DIM + s * KSTEP + part * 8);
        }
        #pragma unroll
        for (int j = 0; j < 2; ++j) {   // B tiles: 512 chunks each
            const int c = tid * 2 + j;
            const int n = c >> 2, part = c & 3;
            const size_t go = (size_t)(n0 + n) * D_DIM + s * KSTEP + part * 8;
            const unsigned int lo = (unsigned int)(n * LDSK + part * 8);
            async_b128(lds_off(&lBx[buf][lo]), Wpx + go);
            async_b128(lds_off(&lBa[buf][lo]), Wpa + go);
        }
    };
    auto compute = [&](int buf) {
        v16bf bx[2], ba[2];
        #pragma unroll
        for (int ni = 0; ni < 2; ++ni) {
            const int row = (wc * 32 + ni * 16 + l16) * LDSK + half * 8;
            bx[ni] = make_frag(&lBx[buf][row], &lBx[buf][row] + 16);
            ba[ni] = make_frag(&lBa[buf][row], &lBa[buf][row] + 16);
        }
        #pragma unroll
        for (int mi = 0; mi < 2; ++mi) {
            const bf16_t* ap = &lA[buf][(wr * 32 + mi * 16 + l16) * LDSK + half * 8];
            v16bf afr = make_frag(ap, ap + 16);
            #pragma unroll
            for (int ni = 0; ni < 2; ++ni) {
                accx[mi][ni] = __builtin_amdgcn_wmma_f32_16x16x32_bf16(
                    false, afr, false, bx[ni], (short)0, accx[mi][ni], false, false);
                acca[mi][ni] = __builtin_amdgcn_wmma_f32_16x16x32_bf16(
                    false, afr, false, ba[ni], (short)0, acca[mi][ni], false, false);
            }
        }
    };

    fill(0, 0);
    wait_async0();
    __syncthreads();
    for (int s = 0; s < NSTG; ++s) {
        const int  cur  = s & 1;
        const bool more = (s + 1 < NSTG);
        if (more) fill(s + 1, cur ^ 1);
        compute(cur);
        wait_async0();
        __syncthreads();
    }

    // ---- fused gate epilogue ----
    #pragma unroll
    for (int ni = 0; ni < 2; ++ni) {
        const int n = n0 + wc * 32 + ni * 16 + l16;
        const float bbx = bgx[n];
        const float bba = bga[n];
        const float ap  = aparam[n];
        const float sp  = log1pf(__expf(ap));   // softplus(a_param)
        #pragma unroll
        for (int mi = 0; mi < 2; ++mi) {
            #pragma unroll
            for (int r = 0; r < 8; ++r) {
                const int m = m0 + wr * 32 + mi * 16 + half * 8 + r;
                const size_t idx = (size_t)m * D_DIM + n;
                const float gx  = accx[mi][ni][r] + bbx;
                const float ga  = acca[mi][ni][r] + bba;
                const float sgx = 1.f / (1.f + __expf(-gx));
                const float sga = 1.f / (1.f + __expf(-ga));
                const float la  = -8.f * sga * sp;
                const float av  = __expf(la);
                const float sc  = sqrtf(fmaxf(1.f - __expf(2.f * la), 0.f));
                const float xv  = (float)X1[idx];
                aOut[idx] = av;
                yOut[idx] = (bf16_t)(sc * sgx * xv);
            }
        }
    }
}

// ===========================================================================
// Sequential diagonal scan: h_t = a_t*h_{t-1} + y_t, per (b,d) channel.
// a in f32 (error compounds), y/h in bf16. Coalesced across d.
// ===========================================================================
__global__ __launch_bounds__(256) void scan_kernel(
    const float* __restrict__ a, const bf16_t* __restrict__ y,
    bf16_t* __restrict__ h)
{
    const int c = blockIdx.x * 256 + threadIdx.x;   // 0..8191 channels
    const int b = c >> 10;                          // / D_DIM
    const int d = c & (D_DIM - 1);
    const size_t base = (size_t)b * S_DIM * D_DIM + d;

    float hv = 0.f;
    for (int t = 0; t < S_DIM; t += 4) {
        const size_t i0 = base + (size_t)t * D_DIM;
        const float a0 = a[i0];
        const float a1 = a[i0 + 1 * D_DIM];
        const float a2 = a[i0 + 2 * D_DIM];
        const float a3 = a[i0 + 3 * D_DIM];
        const float y0 = (float)y[i0];
        const float y1 = (float)y[i0 + 1 * D_DIM];
        const float y2 = (float)y[i0 + 2 * D_DIM];
        const float y3 = (float)y[i0 + 3 * D_DIM];
        hv = fmaf(a0, hv, y0); h[i0]             = (bf16_t)hv;
        hv = fmaf(a1, hv, y1); h[i0 + 1 * D_DIM] = (bf16_t)hv;
        hv = fmaf(a2, hv, y2); h[i0 + 2 * D_DIM] = (bf16_t)hv;
        hv = fmaf(a3, hv, y3); h[i0 + 3 * D_DIM] = (bf16_t)hv;
    }
}

// ===========================================================================
// Launcher
// ===========================================================================
extern "C" void kernel_launch(void* const* d_in, const int* in_sizes, int n_in,
                              void* d_out, int out_size, void* d_ws, size_t ws_size,
                              hipStream_t stream) {
    const float* x       = (const float*)d_in[0];
    const float* W_in    = (const float*)d_in[1];
    const float* b_in    = (const float*)d_in[2];
    const float* W_gx    = (const float*)d_in[3];
    const float* b_gx    = (const float*)d_in[4];
    const float* W_ga    = (const float*)d_in[5];
    const float* b_ga    = (const float*)d_in[6];
    const float* a_param = (const float*)d_in[7];
    const float* W_out   = (const float*)d_in[8];
    const float* b_out   = (const float*)d_in[9];

    const size_t NELEM = (size_t)M_TOT * D_DIM;           // 67,108,864
    const size_t WELEM = (size_t)D_DIM * D_DIM;           // 1,048,576
    char* ws = (char*)d_ws;
    bf16_t* x1    = (bf16_t*)ws;                          // 128 MB
    float*  aArr  = (float*)(ws + NELEM * 2);             // 256 MB
    bf16_t* yArr  = (bf16_t*)(ws + NELEM * 6);            // 128 MB
    bf16_t* hArr  = (bf16_t*)(ws + NELEM * 8);            // 128 MB
    bf16_t* WpIn  = (bf16_t*)(ws + NELEM * 10);           // 2 MB each
    bf16_t* WpGx  = WpIn  + WELEM;
    bf16_t* WpGa  = WpGx  + WELEM;
    bf16_t* WpOut = WpGa  + WELEM;

    dim3 blk(256);
    dim3 pgrid(D_DIM / 32, D_DIM / 32);
    // 0) pack weights: f32 [k][n] -> bf16 [n][k]
    pack_w<<<pgrid, blk, 0, stream>>>(W_in,  WpIn);
    pack_w<<<pgrid, blk, 0, stream>>>(W_gx,  WpGx);
    pack_w<<<pgrid, blk, 0, stream>>>(W_ga,  WpGa);
    pack_w<<<pgrid, blk, 0, stream>>>(W_out, WpOut);
    // 1) x1 = bf16(x @ W_in + b_in)
    gemm_single<0><<<dim3(D_DIM / 128, M_TOT / 128), blk, 0, stream>>>(
        (const void*)x, WpIn, b_in, (void*)x1);
    // 2) gates + a/scale fused
    gemm_dual<<<dim3(D_DIM / 128, M_TOT / 64), blk, 0, stream>>>(
        x1, WpGx, WpGa, b_gx, b_ga, a_param, aArr, yArr);
    // 3) recurrence
    scan_kernel<<<dim3((B_DIM * D_DIM) / 256), blk, 0, stream>>>(aArr, yArr, hArr);
    // 4) out = h @ W_out + b_out
    gemm_single<1><<<dim3(D_DIM / 128, M_TOT / 128), blk, 0, stream>>>(
        (const void*)hArr, WpOut, b_out, d_out);
}
